// LightT2M_678604832988
// MI455X (gfx1250) — matrix operand order repaired and checked by reading, not compiled, not run on hardware
//
#include <hip/hip_runtime.h>
#include <climits>

typedef __attribute__((ext_vector_type(16))) __bf16 v16bf;
typedef __attribute__((ext_vector_type(8)))  float  v8f;
typedef __attribute__((ext_vector_type(4)))  int    v4i;

#define DEV __device__ __forceinline__
#define AS1 __attribute__((address_space(1)))
#define AS3 __attribute__((address_space(3)))

#if defined(__HIP_DEVICE_COMPILE__) && \
    __has_builtin(__builtin_amdgcn_global_load_async_to_lds_b128) && \
    __has_builtin(__builtin_amdgcn_s_wait_asynccnt)
#define GEMM_ASYNC 1
#else
#define GEMM_ASYNC 0
#endif

static constexpr int nB    = 64;
static constexpr int nL    = 196;
static constexpr int nLR   = 197;   // L + 1 token
static constexpr int nD    = 512;
static constexpr int nLp   = 25;    // 200 / 8
static constexpr int nLpad = 200;
static constexpr int nT    = 50;    // 2 * Lp
static constexpr int nDI   = 1024;
static constexpr int nDS   = 16;
static constexpr int nMot  = 263;

// ---------------------------------------------------------------- utilities
DEV int map_row(int r, int q, int s, int rd, int off) {
  return (r / q) * s + (r % q) / rd + off;
}
DEV float sigmoidf_(float x) { return 1.f / (1.f + __expf(-x)); }

// ---------------------------------------------------------------- weight prep
// Wt[n*Kp + k] = bf16( W[k*N + n] ), zero-padded to Kp (multiple of 32)
__global__ void k_wt(__bf16* __restrict__ wt, const float* __restrict__ w,
                     int N, int Ktot, int Kp) {
  long idx = (long)blockIdx.x * 256 + threadIdx.x;
  if (idx >= (long)N * Kp) return;
  int n = (int)(idx / Kp), k = (int)(idx % Kp);
  wt[idx] = (k < Ktot) ? (__bf16)w[(size_t)k * N + n] : (__bf16)0.f;
}

// ---------------------------------------------------------------- WMMA GEMM
// C[M,N] = act( [A0 | A1][M, K0 + rest] @ W[Ktot,N] + bias ), W given as bf16 Wt[N][Kp]
// A frag (16-bit A 16x32): lane&15 = M row; lane>>4 = K+16 block; elements 0..15 -> contiguous K
template <bool KG>
DEV v16bf load_a(const float* b0, const float* b1, int K0, int Ktot, int kk, int kh8) {
  v16bf a;
  if (!KG) {
    const float* base = (kk < K0) ? (b0 + kk) : (b1 + (kk - K0));  // wave-uniform select
    v8f lo = *(const v8f*)(base + kh8);
    v8f hi = *(const v8f*)(base + 8 + kh8);
#pragma unroll
    for (int i = 0; i < 8; ++i) { a[i] = (__bf16)lo[i]; a[i + 8] = (__bf16)hi[i]; }
  } else {
#pragma unroll
    for (int i = 0; i < 8; ++i) {
      int ka = kk + kh8 + i, kb = kk + 8 + kh8 + i;
      float va = (ka < Ktot) ? ((ka < K0) ? b0[ka] : b1[ka - K0]) : 0.f;
      float vb = (kb < Ktot) ? ((kb < K0) ? b0[kb] : b1[kb - K0]) : 0.f;
      a[i] = (__bf16)va; a[i + 8] = (__bf16)vb;
    }
  }
  return a;
}

DEV void store_tile(v8f c, int mbase, int nbase, int lane,
                    const float* bias, float* C, int ldc,
                    int M, int N, int act,
                    int qc, int sc, int rdc, int offc, int cvalid) {
  int n = nbase + (lane & 15);
  if (n >= N) return;
  int moff = (lane >> 4) * 8;
  float bv = bias ? bias[n] : 0.f;
#pragma unroll
  for (int j = 0; j < 8; ++j) {
    int m = mbase + moff + j;
    if (m >= M) continue;
    if ((m % qc) >= cvalid) continue;   // skip padded rows
    int cr = map_row(m, qc, sc, rdc, offc);
    float v = c[j] + bv;
    if (act == 1)      v = fmaxf(v, 0.f);
    else if (act == 2) v = (v > 20.f) ? v : log1pf(__expf(v));  // softplus
    C[(size_t)cr * ldc + n] = v;
  }
}

template <bool KG, bool NG>
__global__ void __launch_bounds__(128)
wmma_gemm_kernel(const float* __restrict__ A0, int lda0, int K0,
                 int q0, int s0, int rd0, int off0,
                 const float* __restrict__ A1, int lda1,
                 int q1, int s1, int rd1, int off1,
                 const __bf16* __restrict__ Wt, const float* __restrict__ bias,
                 float* __restrict__ C, int ldc,
                 int M, int N, int Kp, int Ktot, int act,
                 int qc, int sc, int rdc, int offc, int cvalid) {
  __shared__ alignas(32) __bf16 sB[2][64 * 32];   // double-buffered weight tile
  int tid  = (int)threadIdx.x;
  int wave = tid >> 5;
  int lane = tid & 31;
  int tm  = blockIdx.y * 64 + (wave >> 1) * 32;
  int tnb = blockIdx.x * 64;
  int tn  = tnb + (wave & 1) * 32;
  v8f c00 = {}, c01 = {}, c10 = {}, c11 = {};
  int lr = lane & 15;
  int kh8  = (lane >> 4) << 3;
  int kh16 = (lane >> 4) << 4;
  int m0 = tm + lr, m1 = tm + 16 + lr;
  const float* r00 = A0 + (size_t)map_row(m0, q0, s0, rd0, off0) * lda0;
  const float* r01 = A0 + (size_t)map_row(m1, q0, s0, rd0, off0) * lda0;
  const float* r10 = A1 ? A1 + (size_t)map_row(m0, q1, s1, rd1, off1) * lda1 : nullptr;
  const float* r11 = A1 ? A1 + (size_t)map_row(m1, q1, s1, rd1, off1) * lda1 : nullptr;

  if constexpr (!NG) {
    // -------- cooperative double-buffered LDS staging of the B tile --------
    int scol  = tid >> 1;            // 0..63  column within block tile
    int shalf = (tid & 1) << 4;      // 0 or 16 (k sub-half)
    const __bf16* gstage = Wt + (size_t)(tnb + scol) * Kp + shalf;
    int sidx = scol * 32 + shalf;
    int bcol0 = (wave & 1) * 32 + lr;
    int bcol1 = bcol0 + 16;

    auto stage = [&](int buf, int kk) {
      const __bf16* g = gstage + kk;
      __bf16* l = &sB[buf][sidx];
#if GEMM_ASYNC
      __builtin_amdgcn_global_load_async_to_lds_b128((AS1 v4i*)g, (AS3 v4i*)l, 0, 0);
      __builtin_amdgcn_global_load_async_to_lds_b128((AS1 v4i*)(g + 8), (AS3 v4i*)(l + 8), 0, 0);
#else
      *(v16bf*)l = *(const v16bf*)g;
#endif
    };

    stage(0, 0);
    int buf = 0;
    for (int kk = 0; kk < Kp; kk += 32, buf ^= 1) {
      if (kk + 32 < Kp) {
        stage(buf ^ 1, kk + 32);
#if GEMM_ASYNC
        __builtin_amdgcn_s_wait_asynccnt(2);   // current tile done; next may stay in flight
#endif
      } else {
#if GEMM_ASYNC
        __builtin_amdgcn_s_wait_asynccnt(0);
#endif
      }
      __syncthreads();
      v16bf af0 = load_a<KG>(r00, r10, K0, Ktot, kk, kh16);
      v16bf af1 = load_a<KG>(r01, r11, K0, Ktot, kk, kh16);
      v16bf bf0 = *(const v16bf*)&sB[buf][bcol0 * 32 + kh16];
      v16bf bf1 = *(const v16bf*)&sB[buf][bcol1 * 32 + kh16];
      c00 = __builtin_amdgcn_wmma_f32_16x16x32_bf16(false, af0, false, bf0, (short)0, c00, false, false);
      c01 = __builtin_amdgcn_wmma_f32_16x16x32_bf16(false, af0, false, bf1, (short)0, c01, false, false);
      c10 = __builtin_amdgcn_wmma_f32_16x16x32_bf16(false, af1, false, bf0, (short)0, c10, false, false);
      c11 = __builtin_amdgcn_wmma_f32_16x16x32_bf16(false, af1, false, bf1, (short)0, c11, false, false);
      __syncthreads();                          // protect buffer before reuse
    }
  } else {
    // -------- direct guarded path (N not a multiple of 64: head GEMM) --------
    int c0 = tn + lr, c1 = tn + 16 + lr;
    const __bf16* wb0 = Wt + (size_t)c0 * Kp + kh16;
    const __bf16* wb1 = Wt + (size_t)c1 * Kp + kh16;
    bool c0ok = (c0 < N), c1ok = (c1 < N);
    v16bf bz;
#pragma unroll
    for (int i = 0; i < 16; ++i) bz[i] = (__bf16)0.f;
    for (int kk = 0; kk < Kp; kk += 32) {
      v16bf af0 = load_a<KG>(r00, r10, K0, Ktot, kk, kh16);
      v16bf af1 = load_a<KG>(r01, r11, K0, Ktot, kk, kh16);
      v16bf bf0 = c0ok ? *(const v16bf*)(wb0 + kk) : bz;
      v16bf bf1 = c1ok ? *(const v16bf*)(wb1 + kk) : bz;
      c00 = __builtin_amdgcn_wmma_f32_16x16x32_bf16(false, af0, false, bf0, (short)0, c00, false, false);
      c01 = __builtin_amdgcn_wmma_f32_16x16x32_bf16(false, af0, false, bf1, (short)0, c01, false, false);
      c10 = __builtin_amdgcn_wmma_f32_16x16x32_bf16(false, af1, false, bf0, (short)0, c10, false, false);
      c11 = __builtin_amdgcn_wmma_f32_16x16x32_bf16(false, af1, false, bf1, (short)0, c11, false, false);
    }
  }
  store_tile(c00, tm,      tn,      lane, bias, C, ldc, M, N, act, qc, sc, rdc, offc, cvalid);
  store_tile(c01, tm,      tn + 16, lane, bias, C, ldc, M, N, act, qc, sc, rdc, offc, cvalid);
  store_tile(c10, tm + 16, tn,      lane, bias, C, ldc, M, N, act, qc, sc, rdc, offc, cvalid);
  store_tile(c11, tm + 16, tn + 16, lane, bias, C, ldc, M, N, act, qc, sc, rdc, offc, cvalid);
}

// ---------------------------------------------------------------- elementwise
__global__ void k_mask(float* x, const unsigned char* mask) {
  long idx = (long)blockIdx.x * 256 + threadIdx.x;
  if (idx >= (long)nB * nLR * nD) return;
  int r = (int)(idx / nD);
  int l = r % nLR, bb = r / nLR;
  float m = (l == 0) ? 1.f : (mask[bb * nL + l - 1] ? 1.f : 0.f);
  x[idx] *= m;
}

__global__ void k_pos(float* x) {  // add sinusoid_pos to motion rows
  long idx = (long)blockIdx.x * 256 + threadIdx.x;
  if (idx >= (long)nB * nL * nD) return;
  int d = (int)(idx % nD);
  int r = (int)(idx / nD);
  int l = r % nL, bb = r / nL;
  float f = __expf(-(float)((d >> 1) * 2) * logf(10000.f) / (float)nD);
  float ang = (float)l * f;
  x[((size_t)(bb * nLR + 1 + l)) * nD + d] += (d & 1) ? cosf(ang) : sinf(ang);
}

__global__ void k_temb(float* out, const int* ts) {
  int idx = blockIdx.x * 256 + threadIdx.x;
  if (idx >= nB * nD) return;
  int bb = idx >> 9, j = idx & 511;
  float tv = (float)ts[bb];
  int jj = (j < 256) ? j : j - 256;
  float f = __expf(-logf(10000.f) * (float)jj / 256.f);
  out[idx] = (j < 256) ? cosf(tv * f) : sinf(tv * f);
}

__global__ void k_copy_tok(float* X, const float* TTOK) {
  int idx = blockIdx.x * 256 + threadIdx.x;
  if (idx >= nB * nD) return;
  int bb = idx >> 9, d = idx & 511;
  X[((size_t)bb * nLR) * nD + d] = TTOK[idx];
}

__global__ void k_copy_tok_rows(float* dst, const float* src) {
  int idx = blockIdx.x * 256 + threadIdx.x;
  if (idx >= nB * nD) return;
  int bb = idx >> 9, d = idx & 511;
  dst[((size_t)bb * nLR) * nD + d] = src[((size_t)bb * nLR) * nD + d];
}

__global__ void k_dwconv3(float* out, const float* in, const float* w, const float* bias) {
  long idx = (long)blockIdx.x * 256 + threadIdx.x;
  if (idx >= (long)nB * nLR * nD) return;
  int d = (int)(idx % nD);
  int r = (int)(idx / nD);
  int l = r % nLR, bb = r / nLR;
  float acc = bias[d];
#pragma unroll
  for (int j = 0; j < 3; ++j) {
    int ll = l - 1 + j;
    if (ll >= 0 && ll < nLR) acc += in[((size_t)(bb * nLR + ll)) * nD + d] * w[d * 3 + j];
  }
  out[idx] = acc;
}

__global__ void __launch_bounds__(256)
k_gnorm_relu(float* out, const float* x, const float* g, const float* b) {
  int bg = blockIdx.x;
  int bb = bg >> 4, grp = bg & 15;
  int n = nLR * 32;
  float s = 0.f, s2 = 0.f;
  for (int i = threadIdx.x; i < n; i += 256) {
    int l = i >> 5, c = i & 31;
    float v = x[((size_t)(bb * nLR + l)) * nD + grp * 32 + c];
    s += v; s2 += v * v;
  }
  __shared__ float sh[256], sh2[256];
  sh[threadIdx.x] = s; sh2[threadIdx.x] = s2;
  __syncthreads();
  for (int st = 128; st > 0; st >>= 1) {
    if (threadIdx.x < st) { sh[threadIdx.x] += sh[threadIdx.x + st]; sh2[threadIdx.x] += sh2[threadIdx.x + st]; }
    __syncthreads();
  }
  float mean = sh[0] / (float)n;
  float var  = sh2[0] / (float)n - mean * mean;
  float rstd = rsqrtf(var + 1e-5f);
  for (int i = threadIdx.x; i < n; i += 256) {
    int l = i >> 5, c = i & 31;
    size_t p = ((size_t)(bb * nLR + l)) * nD + grp * 32 + c;
    float v = (x[p] - mean) * rstd * g[grp * 32 + c] + b[grp * 32 + c];
    out[p] = fmaxf(v, 0.f);
  }
}

__global__ void __launch_bounds__(128)
k_ln(float* out, const float* x, const float* res, const float* g, const float* b) {
  int r = blockIdx.x;
  const float* xr = x + (size_t)r * nD;
  const float* rr = res ? res + (size_t)r * nD : nullptr;
  float v[4], s = 0.f, s2 = 0.f;
#pragma unroll
  for (int i = 0; i < 4; ++i) {
    int d = threadIdx.x + i * 128;
    v[i] = xr[d] + (rr ? rr[d] : 0.f);
    s += v[i]; s2 += v[i] * v[i];
  }
  __shared__ float sh[128], sh2[128];
  sh[threadIdx.x] = s; sh2[threadIdx.x] = s2;
  __syncthreads();
  for (int st = 64; st > 0; st >>= 1) {
    if (threadIdx.x < st) { sh[threadIdx.x] += sh[threadIdx.x + st]; sh2[threadIdx.x] += sh2[threadIdx.x + st]; }
    __syncthreads();
  }
  float mean = sh[0] * (1.f / nD);
  float var  = sh2[0] * (1.f / nD) - mean * mean;
  float rstd = rsqrtf(var + 1e-5f);
#pragma unroll
  for (int i = 0; i < 4; ++i) {
    int d = threadIdx.x + i * 128;
    out[(size_t)r * nD + d] = (v[i] - mean) * rstd * g[d] + b[d];
  }
}

__global__ void k_x1p(float* out, const float* X) {
  long idx = (long)blockIdx.x * 256 + threadIdx.x;
  if (idx >= (long)nB * nLpad * nD) return;
  int d = (int)(idx % nD);
  int r = (int)(idx / nD);
  int l = r % nLpad, bb = r / nLpad;
  out[idx] = (l < nL) ? X[((size_t)(bb * nLR + 1 + l)) * nD + d] : 0.f;
}

__global__ void k_nx1(float* out, const float* h, const float* w, const float* bias) {
  long idx = (long)blockIdx.x * 256 + threadIdx.x;
  if (idx >= (long)nB * nLp * nD) return;
  int d = (int)(idx % nD);
  int r = (int)(idx / nD);
  int l = r % nLp, bb = r / nLp;
  float acc = bias[d];
#pragma unroll
  for (int p = 0; p < 8; ++p)
    acc += h[((size_t)(bb * nLpad + l * 8 + p)) * nD + d] * w[d * 8 + p];
  out[idx] = acc;
}

__global__ void k_gated(float* out, const float* Y, const float* yhat) {
  long idx = (long)blockIdx.x * 256 + threadIdx.x;
  if (idx >= (long)nB * nLp * nD) return;
  int d = (int)(idx % nD);
  int bb = (int)(idx / nD) / nLp;
  out[idx] = Y[bb * nD + d] * sigmoidf_(yhat[idx]);
}

__global__ void k_seq(float* out, const float* x2) {
  long idx = (long)blockIdx.x * 256 + threadIdx.x;
  if (idx >= (long)nB * nT * nD) return;
  int d = (int)(idx % nD);
  int r = (int)(idx / nD);
  int t = r % nT, bb = r / nT;
  int src = (t < nLp) ? (nLp - 1 - t) : (t - nLp);
  out[idx] = x2[((size_t)(bb * nLp + src)) * nD + d];
}

__global__ void k_convsilu(float* out, const float* xz, const float* w, const float* bias) {
  long idx = (long)blockIdx.x * 256 + threadIdx.x;
  if (idx >= (long)nB * nT * nDI) return;
  int d = (int)(idx % nDI);
  int r = (int)(idx / nDI);
  int t = r % nT, bb = r / nT;
  float acc = bias[d];
#pragma unroll
  for (int j = 0; j < 4; ++j) {
    int tt = t - 3 + j;
    if (tt >= 0) acc += xz[((size_t)(bb * nT + tt)) * (2 * nDI) + d] * w[d * 4 + j];
  }
  out[idx] = acc * sigmoidf_(acc);
}

__global__ void __launch_bounds__(256)
k_scan(const float* xz, const float* xcs, const float* delta, const float* dbl,
       const float* A_log, const float* Dp, float* yout) {
  int bb = blockIdx.x >> 2;
  int d  = (blockIdx.x & 3) * 256 + threadIdx.x;
  float A[nDS], h[nDS];
#pragma unroll
  for (int s = 0; s < nDS; ++s) { A[s] = -__expf(A_log[d * nDS + s]); h[s] = 0.f; }
  float Dv = Dp[d];
  __shared__ float sB[nDS], sC[nDS];
  for (int t = 0; t < nT; ++t) {
    int row = bb * nT + t;
    if (threadIdx.x < 16)       sB[threadIdx.x]      = dbl[row * 64 + 32 + threadIdx.x];
    else if (threadIdx.x < 32)  sC[threadIdx.x - 16] = dbl[row * 64 + 48 + (threadIdx.x - 16)];
    __syncthreads();
    float dt = delta[(size_t)row * nDI + d];
    float xv = xcs[(size_t)row * nDI + d];
    float dx = dt * xv;
    float acc = 0.f;
#pragma unroll
    for (int s = 0; s < nDS; ++s) {
      h[s] = __expf(dt * A[s]) * h[s] + dx * sB[s];
      acc += h[s] * sC[s];
    }
    float zv = xz[(size_t)row * (2 * nDI) + nDI + d];
    yout[(size_t)row * nDI + d] = (acc + xv * Dv) * (zv * sigmoidf_(zv));
    __syncthreads();
  }
}

// ---------------------------------------------------------------- host side
struct RMap { int q, s, rd, off; };
static const RMap RID = {INT_MAX, 0, 1, 0};

static dim3 ewgrid(long n) { return dim3((unsigned)((n + 255) / 256)); }

static void launch_gemm(hipStream_t st, __bf16* WT,
                        const float* A0, int lda0, int K0, RMap mA0,
                        const float* A1, int lda1, RMap mA1,
                        const float* W, const float* bias,
                        float* C, int ldc, RMap mC, int cvalid,
                        int M, int N, int Ktot, int act) {
  int Kp = (Ktot + 31) & ~31;
  k_wt<<<ewgrid((long)N * Kp), 256, 0, st>>>(WT, W, N, Ktot, Kp);
  dim3 grid((N + 63) / 64, M / 64), blk(128);
  bool kg = (Kp != Ktot), ng = (N & 63) != 0;
  if (!kg && !ng)
    wmma_gemm_kernel<false, false><<<grid, blk, 0, st>>>(
        A0, lda0, K0, mA0.q, mA0.s, mA0.rd, mA0.off, A1, lda1, mA1.q, mA1.s, mA1.rd, mA1.off,
        WT, bias, C, ldc, M, N, Kp, Ktot, act, mC.q, mC.s, mC.rd, mC.off, cvalid);
  else if (kg)
    wmma_gemm_kernel<true, false><<<grid, blk, 0, st>>>(
        A0, lda0, K0, mA0.q, mA0.s, mA0.rd, mA0.off, A1, lda1, mA1.q, mA1.s, mA1.rd, mA1.off,
        WT, bias, C, ldc, M, N, Kp, Ktot, act, mC.q, mC.s, mC.rd, mC.off, cvalid);
  else
    wmma_gemm_kernel<false, true><<<grid, blk, 0, st>>>(
        A0, lda0, K0, mA0.q, mA0.s, mA0.rd, mA0.off, A1, lda1, mA1.q, mA1.s, mA1.rd, mA1.off,
        WT, bias, C, ldc, M, N, Kp, Ktot, act, mC.q, mC.s, mC.rd, mC.off, cvalid);
}

extern "C" void kernel_launch(void* const* d_in, const int* in_sizes, int n_in,
                              void* d_out, int out_size, void* d_ws, size_t ws_size,
                              hipStream_t stream) {
  (void)in_sizes; (void)n_in; (void)out_size; (void)ws_size;
  auto F = [&](int i) { return (const float*)d_in[i]; };

  const float* motion = F(0);
  const unsigned char* mask = (const unsigned char*)d_in[1];
  const int* timestep = (const int*)d_in[2];
  const float* text = F(3);
  // params (jax tree, sorted keys):
  const float* m_in_b  = F(4);
  const float* m_in_w  = F(5);
  const float* m_out_b = F(6);
  const float* m_out_w = F(7);
  const int TIB = 4 + 4 + 39 * 4;   // 164
  const float* t_in_b  = F(TIB + 0);
  const float* t_in_w  = F(TIB + 1);
  const float* time_b  = F(TIB + 2);
  const float* time_w  = F(TIB + 3);

  float* ws = (float*)d_ws;
  size_t off = 0;
  auto alloc = [&](size_t n) { float* p = ws + off; off += n; return p; };
  float* X     = alloc((size_t)nB * nLR * nD);
  float* TEMB  = alloc((size_t)nB * nD);
  float* TTOK  = alloc((size_t)nB * nD);
  float* Y     = alloc((size_t)nB * nD);
  float* H1    = alloc((size_t)nB * nLpad * nD);
  float* H2    = alloc((size_t)nB * nLpad * nD);
  float* X1P   = alloc((size_t)nB * nLpad * nD);
  float* NX1   = alloc((size_t)nB * nLp * nD);
  float* YHAT  = alloc((size_t)nB * nLp * nD);
  float* GATED = alloc((size_t)nB * nLp * nD);
  float* X2    = alloc((size_t)nB * nLp * nD);
  float* SEQ   = alloc((size_t)nB * nT * nD);
  float* XZ    = alloc((size_t)nB * nT * 2 * nDI);
  float* XCS   = alloc((size_t)nB * nT * nDI);
  float* DBL   = alloc((size_t)nB * nT * 64);
  float* DELTA = alloc((size_t)nB * nT * nDI);
  float* MAMY  = alloc((size_t)nB * nT * nDI);
  float* MAMOUT= alloc((size_t)nB * nT * nD);
  float* SEQLN = alloc((size_t)nB * nT * nD);
  __bf16* WT   = (__bf16*)alloc((size_t)524288);  // 1M bf16: max N*Kp = 2048*512

  const long NX_FULL = (long)nB * nLR * nD;

  // ---- embedding: x[:,1:] = motion @ m_in_w + b ; + sinusoid pos
  RMap mMot = {nL, nLR, 1, 1};  // row r -> (r/196)*197 + r%196 + 1
  launch_gemm(stream, WT, motion, nMot, nMot, RID, nullptr, 0, RID,
              m_in_w, m_in_b, X, nD, mMot, INT_MAX, nB * nL, nD, nMot, 0);
  k_pos<<<ewgrid((long)nB * nL * nD), 256, 0, stream>>>(X);
  // time token
  k_temb<<<ewgrid(nB * nD), 256, 0, stream>>>(TEMB, timestep);
  launch_gemm(stream, WT, TEMB, nD, nD, RID, nullptr, 0, RID,
              time_w, time_b, TTOK, nD, RID, INT_MAX, nB, nD, nD, 0);
  k_copy_tok<<<ewgrid(nB * nD), 256, 0, stream>>>(X, TTOK);
  // text token
  launch_gemm(stream, WT, text, nD, nD, RID, nullptr, 0, RID,
              t_in_w, t_in_b, Y, nD, RID, INT_MAX, nB, nD, nD, 0);

  auto run_local = [&](int base) {
    const float* dw_b = F(base + 0); const float* dw_w = F(base + 1);
    const float* gn_b = F(base + 2); const float* gn_g = F(base + 3);
    const float* ln_b = F(base + 4); const float* ln_g = F(base + 5);
    const float* pw_b = F(base + 6); const float* pw_w = F(base + 7);
    k_mask<<<ewgrid(NX_FULL), 256, 0, stream>>>(X, mask);
    launch_gemm(stream, WT, X, nD, nD, RID, nullptr, 0, RID,
                pw_w, pw_b, H1, nD, RID, INT_MAX, nB * nLR, nD, nD, 0);
    k_dwconv3<<<ewgrid(NX_FULL), 256, 0, stream>>>(H2, H1, dw_w, dw_b);
    k_gnorm_relu<<<dim3(nB * 16), 256, 0, stream>>>(H1, H2, gn_g, gn_b);
    k_ln<<<dim3(nB * nLR), 128, 0, stream>>>(X, X, H1, ln_g, ln_b);
  };

  auto run_mixed = [&](int mb) {
    const float* bn_b   = F(mb + 0);  const float* bn_g   = F(mb + 1);
    const float* f_b    = F(mb + 2);  const float* f_w    = F(mb + 3);
    const float* fin_b  = F(mb + 4);  const float* fin_w  = F(mb + 5);
    const float* fuse_b = F(mb + 6);  const float* fuse_w = F(mb + 7);
    const float* lc_dw_b= F(mb + 8);  const float* lc_dw_w= F(mb + 9);
    const float* lc_pw_b= F(mb + 10); const float* lc_pw_w= F(mb + 11);
    const float* ln_b   = F(mb + 12); const float* ln_g   = F(mb + 13);
    const float* A_log  = F(mb + 14); const float* Dp     = F(mb + 15);
    const float* conv_b = F(mb + 16); const float* conv_w = F(mb + 17);
    const float* dt_b   = F(mb + 18); const float* dt_w   = F(mb + 19);
    const float* in_w   = F(mb + 20); const float* out_w  = F(mb + 21);
    const float* xproj_w= F(mb + 22);

    k_mask<<<ewgrid(NX_FULL), 256, 0, stream>>>(X, mask);
    k_x1p<<<ewgrid((long)nB * nLpad * nD), 256, 0, stream>>>(X1P, X);
    launch_gemm(stream, WT, X1P, nD, nD, RID, nullptr, 0, RID,
                lc_pw_w, lc_pw_b, H1, nD, RID, INT_MAX, nB * nLpad, nD, nD, 1);
    k_nx1<<<ewgrid((long)nB * nLp * nD), 256, 0, stream>>>(NX1, H1, lc_dw_w, lc_dw_b);
    RMap mBcast = {nLp, 1, INT_MAX, 0};  // row -> batch index (broadcast y)
    launch_gemm(stream, WT, NX1, nD, nD, RID, Y, nD, mBcast,
                f_w, f_b, YHAT, nD, RID, INT_MAX, nB * nLp, nD, 2 * nD, 0);
    k_gated<<<ewgrid((long)nB * nLp * nD), 256, 0, stream>>>(GATED, Y, YHAT);
    launch_gemm(stream, WT, NX1, nD, nD, RID, GATED, nD, RID,
                fuse_w, fuse_b, X2, nD, RID, INT_MAX, nB * nLp, nD, 2 * nD, 0);
    k_seq<<<ewgrid((long)nB * nT * nD), 256, 0, stream>>>(SEQ, X2);
    // mamba
    launch_gemm(stream, WT, SEQ, nD, nD, RID, nullptr, 0, RID,
                in_w, nullptr, XZ, 2 * nDI, RID, INT_MAX, nB * nT, 2 * nDI, nD, 0);
    k_convsilu<<<ewgrid((long)nB * nT * nDI), 256, 0, stream>>>(XCS, XZ, conv_w, conv_b);
    launch_gemm(stream, WT, XCS, nDI, nDI, RID, nullptr, 0, RID,
                xproj_w, nullptr, DBL, 64, RID, INT_MAX, nB * nT, 64, nDI, 0);
    launch_gemm(stream, WT, DBL, 64, 32, RID, nullptr, 0, RID,   // dt slice, lda=64
                dt_w, dt_b, DELTA, nDI, RID, INT_MAX, nB * nT, nDI, 32, 2);
    k_scan<<<dim3(nB * 4), 256, 0, stream>>>(XZ, XCS, DELTA, DBL, A_log, Dp, MAMY);
    launch_gemm(stream, WT, MAMY, nDI, nDI, RID, nullptr, 0, RID,
                out_w, nullptr, MAMOUT, nD, RID, INT_MAX, nB * nT, nD, nDI, 0);
    k_ln<<<dim3(nB * nT), 128, 0, stream>>>(SEQLN, SEQ, MAMOUT, bn_g, bn_b);
    // fin: concat(x1p, repeat(out[:,Lp:],P)) @ fin_w -> rows 1..196 of H2
    RMap mRep = {nLpad, nT, 8, nLp};   // r -> (r/200)*50 + (r%200)/8 + 25
    RMap mOut = {nLpad, nLR, 1, 1};    // r -> (r/200)*197 + r%200 + 1
    launch_gemm(stream, WT, X1P, nD, nD, RID, SEQLN, nD, mRep,
                fin_w, fin_b, H2, nD, mOut, nL, nB * nLpad, nD, 2 * nD, 0);
    k_copy_tok_rows<<<ewgrid(nB * nD), 256, 0, stream>>>(H2, X);
    k_ln<<<dim3(nB * nLR), 128, 0, stream>>>(X, H2, nullptr, ln_g, ln_b);
  };

  for (int s = 0; s < 4; ++s) {
    int base = 8 + 39 * s;
    run_local(base + 0);       // l1
    run_mixed(base + 16);      // mx
    run_local(base + 8);       // l2
  }

  // ---- head: out = x[:,1:] @ m_out_w + m_out_b
  RMap mSlice = {nL, nLR, 1, 1};
  launch_gemm(stream, WT, X, nD, nD, mSlice, nullptr, 0, RID,
              m_out_w, m_out_b, (float*)d_out, nMot, RID, INT_MAX,
              nB * nL, nMot, nD, 0);
}
